// GATv2Block_1365799600620
// MI455X (gfx1250) — matrix operand order repaired and checked by reading, not compiled, not run on hardware
//
#include <hip/hip_runtime.h>
#include <hip/hip_bf16.h>
#include <math.h>

typedef _Float16 v16h __attribute__((ext_vector_type(16)));
typedef float    v8f  __attribute__((ext_vector_type(8)));
typedef float    v2f  __attribute__((ext_vector_type(2)));

#define BATCH  512
#define CH     256
#define NTOK   81
#define NPAD   96                 // 81 padded to 6x16 tiles
#define CN     (CH * NTOK)        // 20736 elems per batch of x
#define MROWS  (BATCH * NTOK)     // 41472 token rows
#define MTILES (MROWS / 16)       // 2592  (exact)
#define KNN    9
#define EPSV   1e-5f

// ---------------------------------------------------------------------------
// Kernel 1: pack tokens (x in [B,C,H*W]) into f16 A-fragments, pre-swizzled to
// the v_wmma_f32_16x16x32_f16 A-matrix per-lane layout so the GEMM does a
// single contiguous 32B load per fragment.
// ---------------------------------------------------------------------------
__global__ void pack_a(const float* __restrict__ x, _Float16* __restrict__ pA) {
    int e     = blockIdx.x * 256 + threadIdx.x;   // grid sized exactly
    int h     = e & 15;
    int lane  = (e >> 4) & 31;
    int kstep = (e >> 9) & 7;
    int tile  = e >> 12;
    int m     = lane & 15;
    int koff  = (lane >> 4) * 8;
    int j     = h >> 1, p = h & 1;
    int kin   = ((j < 4) ? (2 * j) : (16 + 2 * (j - 4))) + koff + p;
    int k     = kstep * 32 + kin;                 // input channel
    int row   = tile * 16 + m;                    // token row
    int b     = row / NTOK, n = row % NTOK;
    pA[e] = (_Float16)x[(size_t)b * CN + (size_t)k * NTOK + n];
}

// ---------------------------------------------------------------------------
// Kernel 2: pack W^T (B-matrix, [K=c_in][N=c_out]) into f16 B-fragments.
// B layout (32x16 f16): lane L: n = L&15; half h: k = (h<8 ? h : 16+h-8) + (L>>4)*8.
// ---------------------------------------------------------------------------
__global__ void pack_b(const float* __restrict__ wq, const float* __restrict__ wk,
                       const float* __restrict__ wv, _Float16* __restrict__ pB) {
    int e     = blockIdx.x * 256 + threadIdx.x;   // < 3*65536
    int mat   = e >> 16;
    int r     = e & 65535;
    int h     = r & 15;
    int lane  = (r >> 4) & 31;
    int kstep = (r >> 9) & 7;
    int ntile = r >> 12;                          // 0..15
    int n     = ntile * 16 + (lane & 15);         // output channel
    int k     = kstep * 32 + ((h < 8) ? h : (16 + h - 8)) + (lane >> 4) * 8; // input channel
    const float* w = (mat == 0) ? wq : ((mat == 1) ? wk : wv);
    pB[e] = (_Float16)w[(size_t)n * CH + k];      // B[k][n] = W[n][k]
}

// ---------------------------------------------------------------------------
// Kernel 3: per-batch Gram matrix via V_WMMA_F32_16X16X4_F32 into LDS, then
// d^2[n,m] = sq[n] + sq[m] - 2 G[n,m] (sq = Gram diagonal), then 9-NN
// selection. Padded token indices are clamped (EXEC must stay all-ones for
// WMMA); padded rows/cols are never read back.
// f32 A 16x4 layout:  lane L: m = L&15;  elem j: k = j + (L>>4)*2
// f32 B 4x16 layout:  lane L: n = L&15;  elem j: k = (L>>4)*2 + j
// -> identical k pattern, fragments differ only in token index.
// ---------------------------------------------------------------------------
__global__ void knn_kernel(const float* __restrict__ x, int* __restrict__ idx) {
    __shared__ float G[NPAD * NPAD];              // 36864 B
    __shared__ float sq[NPAD];
    int b    = blockIdx.x;
    int tid  = threadIdx.x;
    int lane = tid & 31;
    int wave = tid >> 5;
    const float* xb = x + (size_t)b * CN;

    int mtok16 = lane & 15;                       // row/col within tile
    int khalf  = (lane >> 4) * 2;                 // K sub-offset for this lane

    // 36 tiles over 8 waves: waves 0..3 take 5 tiles, waves 4..7 take 4.
    for (int t = wave; t < 36; t += 8) {
        int ty = t / 6, tx = t % 6;
        int ma = ty * 16 + mtok16;  if (ma > NTOK - 1) ma = NTOK - 1;  // clamp
        int nb = tx * 16 + mtok16;  if (nb > NTOK - 1) nb = NTOK - 1;  // clamp
        v8f acc = {};
        for (int ks = 0; ks < 64; ++ks) {
            int k0 = ks * 4 + khalf;
            v2f af, bf;
            af[0] = xb[(k0    ) * NTOK + ma];
            af[1] = xb[(k0 + 1) * NTOK + ma];
            bf[0] = xb[(k0    ) * NTOK + nb];
            bf[1] = xb[(k0 + 1) * NTOK + nb];
            acc = __builtin_amdgcn_wmma_f32_16x16x4_f32(false, af, false, bf,
                                                        (short)0, acc, false, false);
        }
        // D layout: VGPR r -> (m = r + (lane>=16 ? 8:0), n = lane&15)
        int mhi = (lane >> 4) * 8;
#pragma unroll
        for (int r = 0; r < 8; ++r)
            G[(ty * 16 + mhi + r) * NPAD + tx * 16 + (lane & 15)] = acc[r];
    }
    __syncthreads();
    if (tid < NTOK) sq[tid] = G[tid * NPAD + tid];
    __syncthreads();
    if (tid < NTOK) {
        int n = tid;
        float sqn  = sq[n];
        float* rowp = &G[n * NPAD];               // row n private to this thread
        for (int m = 0; m < NTOK; ++m)
            rowp[m] = sqn + sq[m] - 2.f * rowp[m];
        int base = (b * NTOK + n) * KNN;
        for (int kk = 0; kk < KNN; ++kk) {
            float best = 3.4e38f; int bi = 0;
            for (int m = 0; m < NTOK; ++m) {
                float v = rowp[m];
                if (v < best) { best = v; bi = m; }   // strict < : top_k tie-break
            }
            idx[base + kk] = bi;
            rowp[bi] = 3.4e38f;
        }
    }
}

// ---------------------------------------------------------------------------
// Kernel 4: fused Q/K/V projection GEMM on WMMA (f16 in, f32 accum).
// ---------------------------------------------------------------------------
__global__ void gemm_qkv(const _Float16* __restrict__ pA, const _Float16* __restrict__ pB,
                         const float* __restrict__ bq, const float* __restrict__ bk,
                         const float* __restrict__ bv,
                         float* __restrict__ Q, float* __restrict__ Kf, float* __restrict__ Vf) {
    int lane     = threadIdx.x & 31;
    int wave     = threadIdx.x >> 5;
    int row_tile = blockIdx.x >> 1;
    int n_tile   = ((blockIdx.x & 1) << 3) | wave;   // 0..15
    v8f aq = {}; v8f ak = {}; v8f av = {};
    const _Float16* pBq = pB;
    const _Float16* pBk = pB + 65536;
    const _Float16* pBv = pB + 131072;
#pragma unroll
    for (int ks = 0; ks < 8; ++ks) {
        v16h afrag = *(const v16h*)(pA + (((size_t)row_tile * 8 + ks) * 32 + lane) * 16);
        size_t boff = (((size_t)n_tile * 8 + ks) * 32 + lane) * 16;
        v16h bfq = *(const v16h*)(pBq + boff);
        v16h bfk = *(const v16h*)(pBk + boff);
        v16h bfv = *(const v16h*)(pBv + boff);
        aq = __builtin_amdgcn_wmma_f32_16x16x32_f16(false, afrag, false, bfq, (short)0, aq, false, false);
        ak = __builtin_amdgcn_wmma_f32_16x16x32_f16(false, afrag, false, bfk, (short)0, ak, false, false);
        av = __builtin_amdgcn_wmma_f32_16x16x32_f16(false, afrag, false, bfv, (short)0, av, false, false);
    }
    int col = n_tile * 16 + (lane & 15);
    int mhi = (lane >> 4) * 8;
    float biasq = bq[col], biask = bk[col], biasv = bv[col];
#pragma unroll
    for (int r = 0; r < 8; ++r) {
        int row = row_tile * 16 + mhi + r;
        size_t o = (size_t)row * CH + col;
        Q[o]  = aq[r] + biasq;
        Kf[o] = ak[r] + biask;
        Vf[o] = av[r] + biasv;
    }
}

// ---------------------------------------------------------------------------
// Kernel 5: attention. One wave per token; lane owns 8 contiguous channels,
// so each head (64 ch) = one group of 8 lanes -> shfl_xor 1/2/4 reduction.
// Writes pre-BN output in [B, C, N] layout for the channel reduction.
// ---------------------------------------------------------------------------
__global__ void attn_kernel(const float* __restrict__ Q, const float* __restrict__ Kf,
                            const float* __restrict__ Vf, const int* __restrict__ idx,
                            const float* __restrict__ a, float* __restrict__ outp) {
    int lane = threadIdx.x & 31;
    int wave = threadIdx.x >> 5;
    int row  = blockIdx.x * 8 + wave;             // token row, exact grid
    int b    = row / NTOK, n = row % NTOK;
    int c0   = lane * 8;                          // 8 channels per lane
    float qv[8], aw[8];
#pragma unroll
    for (int i = 0; i < 8; ++i) {
        qv[i] = Q[(size_t)row * CH + c0 + i];
        aw[i] = a[c0 + i];                        // a[h,d] flattened == a[c]
    }
    int   nid[KNN];
    float sc[KNN];
#pragma unroll
    for (int k = 0; k < KNN; ++k) {
        int nb = idx[row * KNN + k];
        nid[k] = nb;
        const float* kr = Kf + ((size_t)b * NTOK + nb) * CH + c0;
        float part = 0.f;
#pragma unroll
        for (int i = 0; i < 8; ++i) {
            float t = qv[i] + kr[i];
            float l = (t > 0.f) ? t : 0.2f * t;   // leaky_relu 0.2
            part += l * aw[i];
        }
        part += __shfl_xor(part, 1, 32);
        part += __shfl_xor(part, 2, 32);
        part += __shfl_xor(part, 4, 32);          // per-head score, 8-lane group
        sc[k] = part;
    }
    float mx = sc[0];
#pragma unroll
    for (int k = 1; k < KNN; ++k) mx = fmaxf(mx, sc[k]);
    float den = 0.f;
#pragma unroll
    for (int k = 0; k < KNN; ++k) { sc[k] = __expf(sc[k] - mx); den += sc[k]; }
    float inv = 1.f / den;
    float acc[8];
#pragma unroll
    for (int i = 0; i < 8; ++i) acc[i] = 0.f;
#pragma unroll
    for (int k = 0; k < KNN; ++k) {
        const float* vr = Vf + ((size_t)b * NTOK + nid[k]) * CH + c0;
        float w = sc[k] * inv;
#pragma unroll
        for (int i = 0; i < 8; ++i) acc[i] += w * vr[i];
    }
#pragma unroll
    for (int i = 0; i < 8; ++i)
        outp[(size_t)b * CN + (size_t)(c0 + i) * NTOK + n] = acc[i];
}

// ---------------------------------------------------------------------------
// Kernel 6: BatchNorm statistics. One block per channel; biased variance.
// ---------------------------------------------------------------------------
__global__ void bn_stats(const float* __restrict__ outp, float* __restrict__ mean,
                         float* __restrict__ istd) {
    __shared__ float ss[256], ss2[256];
    int c = blockIdx.x, tid = threadIdx.x;
    float s = 0.f, s2 = 0.f;
    for (int j = tid; j < MROWS; j += 256) {
        int b = j / NTOK, n = j % NTOK;
        float v = outp[(size_t)b * CN + (size_t)c * NTOK + n];
        s += v; s2 += v * v;
    }
    ss[tid] = s; ss2[tid] = s2;
    __syncthreads();
    for (int st = 128; st > 0; st >>= 1) {
        if (tid < st) { ss[tid] += ss[tid + st]; ss2[tid] += ss2[tid + st]; }
        __syncthreads();
    }
    if (tid == 0) {
        float m   = ss[0] / (float)MROWS;
        float var = ss2[0] / (float)MROWS - m * m;
        mean[c] = m;
        istd[c] = rsqrtf(var + EPSV);
    }
}

// ---------------------------------------------------------------------------
// Kernel 7: BN apply + affine + residual + ReLU.
// ---------------------------------------------------------------------------
__global__ void bn_final(const float* __restrict__ outp, const float* __restrict__ x,
                         const float* __restrict__ mean, const float* __restrict__ istd,
                         const float* __restrict__ gamma, const float* __restrict__ beta,
                         float* __restrict__ out) {
    size_t i = (size_t)blockIdx.x * 256 + threadIdx.x;  // exact grid
    int c = (int)((i % CN) / NTOK);
    float y = (outp[i] - mean[c]) * istd[c] * gamma[c] + beta[c] + x[i];
    out[i] = fmaxf(y, 0.f);
}

// ---------------------------------------------------------------------------
extern "C" void kernel_launch(void* const* d_in, const int* in_sizes, int n_in,
                              void* d_out, int out_size, void* d_ws, size_t ws_size,
                              hipStream_t stream) {
    const float* x     = (const float*)d_in[0];
    const float* wq    = (const float*)d_in[1];
    const float* bq    = (const float*)d_in[2];
    const float* wk    = (const float*)d_in[3];
    const float* bk    = (const float*)d_in[4];
    const float* wv    = (const float*)d_in[5];
    const float* bv    = (const float*)d_in[6];
    const float* a     = (const float*)d_in[7];
    const float* gamma = (const float*)d_in[8];
    const float* beta  = (const float*)d_in[9];
    float* out = (float*)d_out;

    // workspace layout (all offsets 32B-aligned)
    char* ws = (char*)d_ws;
    const size_t PACKA_BYTES = (size_t)MROWS * CH * 2;          // 21,233,664
    const size_t PACKB_BYTES = (size_t)3 * CH * CH * 2;         //    393,216
    const size_t QKV_BYTES   = (size_t)MROWS * CH * 4;          // 42,467,328
    const size_t IDX_BYTES   = (size_t)MROWS * KNN * 4;         //  1,492,992
    size_t off = 0;
    _Float16* pA   = (_Float16*)(ws + off); off += PACKA_BYTES;
    _Float16* pB   = (_Float16*)(ws + off); off += PACKB_BYTES;
    float*    Q    = (float*)(ws + off);    off += QKV_BYTES;
    float*    Kf   = (float*)(ws + off);    off += QKV_BYTES;
    float*    Vf   = (float*)(ws + off);    off += QKV_BYTES;
    int*      idx  = (int*)(ws + off);      off += IDX_BYTES;
    float*    pre  = (float*)(ws + off);    off += QKV_BYTES;
    float*    mean = (float*)(ws + off);    off += 1024;
    float*    istd = (float*)(ws + off);    off += 1024;

    pack_a<<<dim3(MROWS * CH / 256), dim3(256), 0, stream>>>(x, pA);
    pack_b<<<dim3(3 * CH * CH / 256), dim3(256), 0, stream>>>(wq, wk, wv, pB);
    knn_kernel<<<dim3(BATCH), dim3(256), 0, stream>>>(x, idx);
    gemm_qkv<<<dim3(MTILES * 2), dim3(256), 0, stream>>>(pA, pB, bq, bk, bv, Q, Kf, Vf);
    attn_kernel<<<dim3(MROWS / 8), dim3(256), 0, stream>>>(Q, Kf, Vf, idx, a, pre);
    bn_stats<<<dim3(CH), dim3(256), 0, stream>>>(pre, mean, istd);
    bn_final<<<dim3((int)((size_t)BATCH * CN / 256), 1, 1), dim3(256), 0, stream>>>(
        pre, x, mean, istd, gamma, beta, out);
}
